// VolumeRenderer_35527969472978
// MI455X (gfx1250) — matrix operand order repaired and checked by reading, not compiled, not run on hardware
//
#include <hip/hip_runtime.h>

typedef __attribute__((ext_vector_type(16))) _Float16 v16h;
typedef __attribute__((ext_vector_type(8)))  _Float16 v8h;
typedef __attribute__((ext_vector_type(8)))  float    v8f;

#define B_RAYS 4096
#define K_SAMP 256
#define D_EMB  32
#define NSAMP  (B_RAYS * K_SAMP)

// ---- WMMA fragment loaders (CDNA5 16-bit layouts, wave32) -------------------
// A 16x32 f16: lane l -> M = l&15, halfsel = l>>4;
//   elements e=0..7  : K = hs*8 + e          (contiguous 16B)
//   elements e=8..15 : K = 16 + hs*8 + (e-8) (contiguous 16B)
__device__ __forceinline__ v16h load_a_frag(const _Float16* tile, int lane, int kbase) {
    int M  = lane & 15;
    int hs = lane >> 4;
    const v8h* p0 = (const v8h*)(tile + M * 64 + kbase + hs * 8);
    const v8h* p1 = (const v8h*)(tile + M * 64 + kbase + 16 + hs * 8);
    v8h lo = *p0, hi = *p1;
    return __builtin_shufflevector(lo, hi, 0,1,2,3,4,5,6,7,8,9,10,11,12,13,14,15);
}
// B 32x16 f16 from [n][k]-transposed weights: lane l -> N = l&15,
//   element e<8 -> K = (l>>4)*8 + e; e>=8 -> K = 16 + (l>>4)*8 + (e-8)  (two 16B)
__device__ __forceinline__ v16h load_b_frag(const _Float16* wt, int lane, int ntile, int kbase) {
    int n  = ntile * 16 + (lane & 15);
    int hs = lane >> 4;
    const v8h* p = (const v8h*)(wt + n * 64 + kbase + hs * 8);
    v8h lo = p[0], hi = p[2];
    return __builtin_shufflevector(lo, hi, 0,1,2,3,4,5,6,7,8,9,10,11,12,13,14,15);
}

__device__ __forceinline__ v8f wmma_f16(v16h a, v16h b, v8f c) {
    return __builtin_amdgcn_wmma_f32_16x16x32_f16(false, a, false, b, (short)0, c, false, false);
}

// ---- Kernel 1: per-sample MLP via WMMA -------------------------------------
__global__ __launch_bounds__(256) void nerf_mlp_kernel(
    const float* __restrict__ ray_start, const float* __restrict__ ray_dir,
    const float* __restrict__ sdepth,    const int*   __restrict__ sidx,
    const float* __restrict__ emb,
    const float* __restrict__ W1, const float* __restrict__ b1,
    const float* __restrict__ W2, const float* __restrict__ b2,
    const float* __restrict__ Wout, const float* __restrict__ bout,
    float* __restrict__ sig_out, float* __restrict__ tex_out)
{
    __shared__ __align__(16) _Float16 w1t[64 * 64];   // [n][k], rows k>=38 zero
    __shared__ __align__(16) _Float16 w2t[64 * 64];   // [n][k]
    __shared__ __align__(16) _Float16 w3t[16 * 64];   // [n][k], n>=4 zero
    __shared__ float bias1[64], bias2[64], bias3[16];
    __shared__ __align__(16) _Float16 featT[8][16 * 64];  // per-wave 16x64 tile
    __shared__ __align__(16) _Float16 hT[8][16 * 64];

    const int tid = threadIdx.x;
    // Preload + transpose weights to f16 (B-matrix friendly [n][k] layout).
    for (int i = tid; i < 64 * 64; i += 256) {
        int n = i >> 6, k = i & 63;
        w1t[i] = (k < 38) ? (_Float16)W1[k * 64 + n] : (_Float16)0.f;
        w2t[i] = (_Float16)W2[k * 64 + n];
    }
    for (int i = tid; i < 16 * 64; i += 256) {
        int n = i >> 6, k = i & 63;
        w3t[i] = (n < 4) ? (_Float16)Wout[k * 4 + n] : (_Float16)0.f;
    }
    if (tid < 64) { bias1[tid] = b1[tid]; bias2[tid] = b2[tid]; }
    if (tid < 16) { bias3[tid] = (tid < 4) ? bout[tid] : 0.f; }

    const int wave = tid >> 5;
    const int lane = tid & 31;
    _Float16* feat = featT[wave];
    _Float16* hbuf = hT[wave];
    const int tileBase = blockIdx.x * 128 + wave * 16;  // 8 waves * 16 samples

    // Build feature tile: [16 samples][64 cols f16]; cols = pts(3)|emb(32)|dirs(3)|pad
    {
        int s     = lane >> 1;           // 2 lanes per sample
        int cbase = (lane & 1) * 32;     // each lane fills 32 cols
        int sample = tileBase + s;
        int bray  = sample >> 8;         // K = 256
        float dep = sdepth[sample];
        int iv    = sidx[sample];
        int safe  = iv < 0 ? 0 : iv;
        const float* er = emb + (long)safe * D_EMB;
        float dir[3], org[3];
        #pragma unroll
        for (int j = 0; j < 3; ++j) {
            dir[j] = ray_dir[bray * 3 + j];
            org[j] = ray_start[bray * 3 + j] + dir[j] * dep;
        }
        for (int c = cbase; c < cbase + 32; ++c) {
            float v;
            if      (c < 3)  v = org[c];
            else if (c < 35) v = er[c - 3];
            else if (c < 38) v = dir[c - 35];
            else             v = 0.f;
            feat[s * 64 + c] = (_Float16)v;
        }
    }
    __syncthreads();

    const int N  = lane & 15;
    const int Mb = (lane >> 4) * 8;

    // Layer 1: feat(16x64) @ W1(64x64) -> hbuf, relu
    #pragma unroll
    for (int nt = 0; nt < 4; ++nt) {
        v8f acc = {};
        #pragma unroll
        for (int ks = 0; ks < 2; ++ks)
            acc = wmma_f16(load_a_frag(feat, lane, ks * 32),
                           load_b_frag(w1t, lane, nt, ks * 32), acc);
        float bv = bias1[nt * 16 + N];
        #pragma unroll
        for (int r = 0; r < 8; ++r) {
            float v = acc[r] + bv;
            hbuf[(Mb + r) * 64 + nt * 16 + N] = (_Float16)(v > 0.f ? v : 0.f);
        }
    }
    __syncthreads();

    // Layer 2: hbuf @ W2 -> feat (reuse), relu
    #pragma unroll
    for (int nt = 0; nt < 4; ++nt) {
        v8f acc = {};
        #pragma unroll
        for (int ks = 0; ks < 2; ++ks)
            acc = wmma_f16(load_a_frag(hbuf, lane, ks * 32),
                           load_b_frag(w2t, lane, nt, ks * 32), acc);
        float bv = bias2[nt * 16 + N];
        #pragma unroll
        for (int r = 0; r < 8; ++r) {
            float v = acc[r] + bv;
            feat[(Mb + r) * 64 + nt * 16 + N] = (_Float16)(v > 0.f ? v : 0.f);
        }
    }
    __syncthreads();

    // Layer 3: feat @ Wout(64x4 padded to 16) -> sigma + tex
    {
        v8f acc = {};
        #pragma unroll
        for (int ks = 0; ks < 2; ++ks)
            acc = wmma_f16(load_a_frag(feat, lane, ks * 32),
                           load_b_frag(w3t, lane, 0, ks * 32), acc);
        if (N < 4) {
            float bv = bias3[N];
            #pragma unroll
            for (int r = 0; r < 8; ++r) {
                int sample = tileBase + Mb + r;
                float v = acc[r] + bv;
                if (N == 0) sig_out[sample] = v;
                else        tex_out[(long)sample * 3 + (N - 1)] = v;
            }
        }
    }
}

// ---- Kernel 2: per-ray alpha compositing scan (one wave per ray) -----------
__global__ __launch_bounds__(256) void nerf_render_kernel(
    const float* __restrict__ sdepth, const int* __restrict__ sidx,
    const float* __restrict__ sdists,
    const float* __restrict__ sig, const float* __restrict__ tex,
    float* __restrict__ colors, float* __restrict__ depths,
    float* __restrict__ missed, float* __restrict__ probs)
{
    const int wave = threadIdx.x >> 5;
    const int lane = threadIdx.x & 31;
    const int ray  = blockIdx.x * 8 + wave;
    const long base = (long)ray * K_SAMP;

    float carry = 0.f, psum = 0.f, dsum = 0.f, c0 = 0.f, c1 = 0.f, c2 = 0.f;
    for (int ch = 0; ch < 8; ++ch) {
        long k = base + ch * 32 + lane;
        float sg = sig[k];
        float fe = (sidx[k] >= 0) ? fmaxf(sg, 0.f) * sdists[k] : 0.f;
        // inclusive wave scan of fe
        float scan = fe;
        #pragma unroll
        for (int off = 1; off < 32; off <<= 1) {
            float nv = __shfl_up(scan, off, 32);
            if (lane >= off) scan += nv;
        }
        float excl = carry + scan - fe;       // exclusive prefix across ray
        float a  = 1.f - __expf(-fe);
        float tr = __expf(-excl);
        float p  = a * tr;
        probs[k] = p;
        psum += p;
        dsum += sdepth[k] * p;
        c0 += tex[k * 3 + 0] * p;   // p==0 wherever mask is false -> matches ref
        c1 += tex[k * 3 + 1] * p;
        c2 += tex[k * 3 + 2] * p;
        carry += __shfl(scan, 31, 32);
    }
    #pragma unroll
    for (int off = 16; off; off >>= 1) {
        psum += __shfl_down(psum, off, 32);
        dsum += __shfl_down(dsum, off, 32);
        c0   += __shfl_down(c0, off, 32);
        c1   += __shfl_down(c1, off, 32);
        c2   += __shfl_down(c2, off, 32);
    }
    if (lane == 0) {
        colors[ray * 3 + 0] = c0;
        colors[ray * 3 + 1] = c1;
        colors[ray * 3 + 2] = c2;
        depths[ray] = dsum;
        missed[ray] = 1.f - psum;
    }
}

extern "C" void kernel_launch(void* const* d_in, const int* in_sizes, int n_in,
                              void* d_out, int out_size, void* d_ws, size_t ws_size,
                              hipStream_t stream) {
    const float* ray_start = (const float*)d_in[0];
    const float* ray_dir   = (const float*)d_in[1];
    const float* sdepth    = (const float*)d_in[2];
    const int*   sidx      = (const int*)  d_in[3];
    const float* sdists    = (const float*)d_in[4];
    const float* emb       = (const float*)d_in[5];
    const float* W1   = (const float*)d_in[6];
    const float* b1   = (const float*)d_in[7];
    const float* W2   = (const float*)d_in[8];
    const float* b2   = (const float*)d_in[9];
    const float* Wout = (const float*)d_in[10];
    const float* bout = (const float*)d_in[11];

    float* sig = (float*)d_ws;               // NSAMP floats
    float* tex = sig + (size_t)NSAMP;        // 3*NSAMP floats

    float* out    = (float*)d_out;
    float* colors = out;                      // B*3
    float* depths = out + (size_t)B_RAYS * 3; // B
    float* missed = depths + B_RAYS;          // B
    float* probs  = missed + B_RAYS;          // B*K

    nerf_mlp_kernel<<<NSAMP / 128, 256, 0, stream>>>(
        ray_start, ray_dir, sdepth, sidx, emb,
        W1, b1, W2, b2, Wout, bout, sig, tex);

    nerf_render_kernel<<<B_RAYS / 8, 256, 0, stream>>>(
        sdepth, sidx, sdists, sig, tex, colors, depths, missed, probs);
}
